// ProposedModel_4346506903830
// MI455X (gfx1250) — compile-verified, matrix operand order checked by reading, not run on hardware
//
#include <hip/hip_runtime.h>

// ---------------- types / WMMA helpers (CDNA5 gfx1250, wave32) ----------------
typedef __bf16 bf16;
typedef __bf16 v16bf __attribute__((ext_vector_type(16)));
typedef float  v8f   __attribute__((ext_vector_type(8)));

union FragU { v16bf v; unsigned u[8]; };
union Pk4   { bf16 h[4]; uint2 u2; };

// Load a 16x32 bf16 fragment (A layout per ISA 7.12.2; also valid for the
// B operand when B^T (= W[N,K]) is stored row-major with even row stride).
// lane l<16 : row l,    K pairs {0..7} in u[0..3], {16..23} in u[4..7]
// lane l>=16: row l-16, K pairs {8..15} in u[0..3], {24..31} in u[4..7]
__device__ __forceinline__ v16bf frag_load(const bf16* p, int row_stride) {
  int lane = threadIdx.x & 31;
  int r = lane & 15, hi = lane >> 4;
  const unsigned* q = (const unsigned*)(p + (long long)r * row_stride);
  FragU f;
  int b0 = hi * 4;
#pragma unroll
  for (int j = 0; j < 4; ++j) { f.u[j] = q[b0 + j]; f.u[4 + j] = q[8 + b0 + j]; }
  return f.v;
}

// ---------------- generic WMMA GEMM: C = alpha * A @ B^T + bias --------------
// A: f32 [M,K] (lda, batch stride sA), converted to bf16 while staging.
// BMODE 0: B = bf16 [N,K] row-major (pre-converted weights)
// BMODE 1: B = f32  [N,K] row-major (Q.K^T style, converted while staging)
// BMODE 2: B = f32  [K,N] row-major (attn.V style, transposed while staging)
// Tile: 64(M) x 128(N) x 32(K), 256 threads = 8 waves, each wave 4 acc tiles.
// Software-pipelined: chunk k+1 global loads overlap chunk k WMMAs.
template <int BMODE>
__global__ __launch_bounds__(256)
void wmma_gemm(const float* __restrict__ A, int lda, long long sA,
               const void* __restrict__ Bv, int ldb, long long sB,
               const float* __restrict__ bias,
               float* __restrict__ C, int ldc, long long sC,
               int M, int N, int K, float alpha)
{
  __shared__ __align__(16) bf16 As[64 * 32];
  __shared__ __align__(16) bf16 Ws[128 * 32];
  int bz = blockIdx.z;
  const float* Ab = A + (long long)bz * sA;
  float* Cb = C + (long long)bz * sC;
  int m0 = blockIdx.x * 64;
  int n0 = blockIdx.y * 128;
  int tid = threadIdx.x;
  int w = tid >> 5;
  int msub = w & 3;       // which 16-row strip of the 64-row tile
  int nhalf = w >> 2;     // which 64-col half of the 128-col tile
  int nk = K / 32;

  v8f acc[4] = {};

  float4 aR[2];           // A: 512 float4 per 64x32 chunk -> 2/thread
  uint4  bR0[2];          // B mode0: 512 uint4 (8 bf16) -> 2/thread
  float4 bR1[4];          // B mode1: 1024 float4 -> 4/thread
  float  bR2[16];         // B mode2: 4096 f32 -> 16/thread

  auto loadA = [&](int k0) {
#pragma unroll
    for (int i = 0; i < 2; ++i) {
      int idx = tid + i * 256;            // 512 float4
      int r = idx >> 3, c4 = idx & 7;
      aR[i] = *(const float4*)(Ab + (long long)(m0 + r) * lda + k0 + c4 * 4);
    }
  };
  auto storeA = [&]() {
#pragma unroll
    for (int i = 0; i < 2; ++i) {
      int idx = tid + i * 256;
      int r = idx >> 3, c4 = idx & 7;
      Pk4 p;
      p.h[0] = (bf16)aR[i].x; p.h[1] = (bf16)aR[i].y;
      p.h[2] = (bf16)aR[i].z; p.h[3] = (bf16)aR[i].w;
      *(uint2*)(As + r * 32 + c4 * 4) = p.u2;
    }
  };
  auto loadB = [&](int k0) {
    if (BMODE == 0) {
      const bf16* Bb = (const bf16*)Bv + (long long)bz * sB;
#pragma unroll
      for (int i = 0; i < 2; ++i) {
        int idx = tid + i * 256;          // 512 uint4 (8 bf16 each)
        int r = idx >> 2, c8 = idx & 3;
        bR0[i] = *(const uint4*)(Bb + (long long)(n0 + r) * ldb + k0 + c8 * 8);
      }
    } else if (BMODE == 1) {
      const float* Bb = (const float*)Bv + (long long)bz * sB;
#pragma unroll
      for (int i = 0; i < 4; ++i) {
        int idx = tid + i * 256;          // 1024 float4
        int r = idx >> 3, c4 = idx & 7;
        bR1[i] = *(const float4*)(Bb + (long long)(n0 + r) * ldb + k0 + c4 * 4);
      }
    } else {
      const float* Bb = (const float*)Bv + (long long)bz * sB; // [K,N] rows
#pragma unroll
      for (int i = 0; i < 16; ++i) {
        int idx = tid + i * 256;
        int n = idx & 127, kk = idx >> 7;
        bR2[i] = Bb[(long long)(k0 + kk) * ldb + (n0 + n)];
      }
    }
  };
  auto storeB = [&]() {
    if (BMODE == 0) {
#pragma unroll
      for (int i = 0; i < 2; ++i) {
        int idx = tid + i * 256;
        int r = idx >> 2, c8 = idx & 3;
        *(uint4*)(Ws + r * 32 + c8 * 8) = bR0[i];
      }
    } else if (BMODE == 1) {
#pragma unroll
      for (int i = 0; i < 4; ++i) {
        int idx = tid + i * 256;
        int r = idx >> 3, c4 = idx & 7;
        Pk4 p;
        p.h[0] = (bf16)bR1[i].x; p.h[1] = (bf16)bR1[i].y;
        p.h[2] = (bf16)bR1[i].z; p.h[3] = (bf16)bR1[i].w;
        *(uint2*)(Ws + r * 32 + c4 * 4) = p.u2;
      }
    } else {
#pragma unroll
      for (int i = 0; i < 16; ++i) {
        int idx = tid + i * 256;
        int n = idx & 127, kk = idx >> 7;
        Ws[n * 32 + kk] = (bf16)bR2[i];
      }
    }
  };

  loadA(0);
  loadB(0);
  for (int kt = 0; kt < nk; ++kt) {
    storeA();
    storeB();
    __syncthreads();
    if (kt + 1 < nk) {        // next chunk's global loads overlap the WMMAs
      loadA((kt + 1) * 32);
      loadB((kt + 1) * 32);
    }
    v16bf a = frag_load(As + msub * 16 * 32, 32);
#pragma unroll
    for (int i = 0; i < 4; ++i) {
      v16bf b = frag_load(Ws + (nhalf * 4 + i) * 16 * 32, 32);
      acc[i] = __builtin_amdgcn_wmma_f32_16x16x32_bf16(
          false, a, false, b, (short)0, acc[i], false, false);
    }
    __syncthreads();
  }
  // ---- store D (+bias, *alpha); C/D layout per ISA 7.12.2 ----
  int lane = tid & 31;
  int col = lane & 15, hi = lane >> 4;
#pragma unroll
  for (int i = 0; i < 4; ++i) {
    int nt = n0 + (nhalf * 4 + i) * 16 + col;
    float bv = bias ? bias[nt] : 0.f;
#pragma unroll
    for (int r = 0; r < 8; ++r) {
      int mrow = m0 + msub * 16 + r + hi * 8;
      Cb[(long long)mrow * ldc + nt] = alpha * acc[i][r] + bv;
    }
  }
}

// ---------------- recurrent GRU scan (16 batches / workgroup) ----------------
// xg [Btot,S,768] f32 (already includes b_ih), Whh bf16 [768,256], bhh f32[768]
// out [Btot,S,256]. Per step: gh = h @ Whh^T (WMMA, M=16,N=768,K=256), then
// fused gate math. h kept f32 + bf16 mirror in LDS. (Codegen note: compiler
// keeps all 48 Whh fragments register-resident across steps via VGPR-MSB.)
__global__ __launch_bounds__(256)
void gru_scan(const float* __restrict__ xg, const bf16* __restrict__ Whh,
              const float* __restrict__ bhh_g, float* __restrict__ out,
              int S, int Btot)
{
  extern __shared__ char smem[];
  float* hF  = (float*)smem;                                   // 16*256 f32
  bf16*  hB  = (bf16*)(smem + 16 * 256 * 4);                   // 16*256 bf16
  float* gh  = (float*)(smem + 16 * 256 * 4 + 16 * 256 * 2);   // 16*768 f32
  float* bsh = gh + 16 * 768;                                  // 768 f32
  int tid = threadIdx.x;
  int b0 = blockIdx.x * 16;
  int w = tid >> 5;

#pragma unroll
  for (int i = 0; i < 16; ++i) { int e = tid + i * 256; hF[e] = 0.f; hB[e] = (bf16)0.f; }
  for (int e = tid; e < 768; e += 256) bsh[e] = bhh_g[e];
  __syncthreads();

  for (int t = 0; t < S; ++t) {
    // gh = h @ Whh^T : 48 N-tiles, 6 per wave, K = 8 chunks of 32
#pragma unroll
    for (int i = 0; i < 6; ++i) {
      int n0 = (w * 6 + i) * 16;
      v8f acc = {};
#pragma unroll
      for (int kc = 0; kc < 8; ++kc) {
        v16bf a  = frag_load(hB + kc * 32, 256);
        v16bf bb = frag_load(Whh + (long long)n0 * 256 + kc * 32, 256);
        acc = __builtin_amdgcn_wmma_f32_16x16x32_bf16(
            false, a, false, bb, (short)0, acc, false, false);
      }
      int lane = tid & 31;
      int col = lane & 15, hi2 = lane >> 4;
#pragma unroll
      for (int r = 0; r < 8; ++r)
        gh[(r + hi2 * 8) * 768 + n0 + col] = acc[r];
    }
    __syncthreads();
    // fused gate math (one (batch,hidden) element per thread-iter)
#pragma unroll
    for (int i = 0; i < 16; ++i) {
      int e = tid + i * 256;
      int j = e & 255, bi = e >> 8;
      long long base = ((long long)(b0 + bi) * S + t) * 768;
      float xr = xg[base + j], xz = xg[base + 256 + j], xn = xg[base + 512 + j];
      float hr = gh[bi * 768 + j]        + bsh[j];
      float hz = gh[bi * 768 + 256 + j]  + bsh[256 + j];
      float hn = gh[bi * 768 + 512 + j]  + bsh[512 + j];
      float rr = 1.f / (1.f + __expf(-(xr + hr)));
      float zz = 1.f / (1.f + __expf(-(xz + hz)));
      float nn = tanhf(xn + rr * hn);
      float hnew = (1.f - zz) * nn + zz * hF[e];
      hF[e] = hnew; hB[e] = (bf16)hnew;
      out[((long long)(b0 + bi) * S + t) * 256 + j] = hnew;
    }
    __syncthreads();
  }
}

// ---------------- small utility kernels ----------------
__global__ void cvt_bf16(const float* __restrict__ s, bf16* __restrict__ d, int n) {
  int i = blockIdx.x * blockDim.x + threadIdx.x;
  if (i < n) d[i] = (bf16)s[i];
}

__global__ void fill0(float* p, long long n) {
  long long i = (long long)blockIdx.x * blockDim.x + threadIdx.x;
  if (i < n) p[i] = 0.f;
}

// Stable gather of rows where (speaker==sid && s<len) to front; comp pre-zeroed.
__global__ void compact_kernel(const float* __restrict__ src, const int* __restrict__ lengths,
                               const int* __restrict__ spk, int sid,
                               float* __restrict__ comp, int* __restrict__ neff)
{
  __shared__ int pos[512];
  int b = blockIdx.x, t = threadIdx.x;
  if (t == 0) {
    int cnt = 0, len = lengths[b];
    for (int s = 0; s < 512; ++s) {
      bool ok = (spk[b * 512 + s] == sid) && (s < len);
      pos[s] = ok ? cnt : -1;
      if (ok) cnt++;
    }
    neff[b] = cnt > 0 ? cnt : 1;
  }
  __syncthreads();
  for (int s = 0; s < 512; ++s) {
    int p = pos[s];
    if (p >= 0)
      comp[((long long)b * 512 + p) * 256 + t] = src[((long long)b * 512 + s) * 256 + t];
  }
}

// Prefix-masked softmax over rows of 512; one wave per row; zero tail.
__global__ __launch_bounds__(256)
void softmax_rows(float* __restrict__ sc, const int* __restrict__ neff, int S)
{
  int row = blockIdx.x * 8 + (threadIdx.x >> 5);
  int lane = threadIdx.x & 31;
  int klen = neff[row / S];
  float* p = sc + (long long)row * S;
  float v[16];
  float mx = -3.4e38f;
#pragma unroll
  for (int i = 0; i < 16; ++i) {
    int c = lane + i * 32;
    v[i] = (c < klen) ? p[c] : -3.4e38f;
    mx = fmaxf(mx, v[i]);
  }
#pragma unroll
  for (int o = 16; o > 0; o >>= 1) mx = fmaxf(mx, __shfl_xor(mx, o));
  float sum = 0.f;
#pragma unroll
  for (int i = 0; i < 16; ++i) {
    int c = lane + i * 32;
    v[i] = (c < klen) ? __expf(v[i] - mx) : 0.f;
    sum += v[i];
  }
#pragma unroll
  for (int o = 16; o > 0; o >>= 1) sum += __shfl_xor(sum, o);
  float inv = 1.f / sum;
#pragma unroll
  for (int i = 0; i < 16; ++i) p[lane + i * 32] = v[i] * inv;
}

// Masked mean over valid query rows: hout[b,h] = sum_{q<neff} ctx[b,q,h] / neff
__global__ void pool_kernel(const float* __restrict__ ctx, const int* __restrict__ neff,
                            float* __restrict__ hout)
{
  int b = blockIdx.x, h = threadIdx.x;
  int n = neff[b];
  float acc = 0.f;
  for (int q = 0; q < n; ++q) acc += ctx[((long long)b * 512 + q) * 256 + h];
  hout[b * 256 + h] = acc / (float)n;
}

// out[b,c] = [hS|hI|hA] . fc_w[c] + fc_b[c];  hA = out1[b, len-1]
__global__ __launch_bounds__(128)
void fc_kernel(const float* __restrict__ hS, const float* __restrict__ hI,
               const float* __restrict__ out1, const int* __restrict__ lengths,
               const float* __restrict__ fw, const float* __restrict__ fb,
               float* __restrict__ out)
{
  int b = blockIdx.x;
  int c = threadIdx.x >> 5;
  int lane = threadIdx.x & 31;
  const float* hA = out1 + ((long long)b * 512 + (lengths[b] - 1)) * 256;
  float acc = 0.f;
  for (int f = lane; f < 768; f += 32) {
    float hv = (f < 256) ? hS[b * 256 + f]
             : (f < 512) ? hI[b * 256 + (f - 256)]
                         : hA[f - 512];
    acc += hv * fw[c * 768 + f];
  }
#pragma unroll
  for (int o = 16; o > 0; o >>= 1) acc += __shfl_xor(acc, o);
  if (lane == 0) out[b * 4 + c] = acc + fb[c];
}

// ---------------- host orchestration ----------------
extern "C" void kernel_launch(void* const* d_in, const int* in_sizes, int n_in,
                              void* d_out, int out_size, void* d_ws, size_t ws_size,
                              hipStream_t stream)
{
  (void)in_sizes; (void)n_in; (void)out_size; (void)ws_size;
  const float* x = (const float*)d_in[0];
  const int M = 64 * 512;           // B*S rows

  char* ws = (char*)d_ws;
  size_t off = 0;
  auto alloc = [&](size_t bytes) -> void* {
    void* p = ws + off;
    off = (off + bytes + 255) & ~(size_t)255;
    return p;
  };

  // bf16 weight copies
  bf16* wGih0 = (bf16*)alloc((size_t)768 * 1024 * 2);
  bf16* wGhh0 = (bf16*)alloc((size_t)768 * 256 * 2);
  bf16* wGih1 = (bf16*)alloc((size_t)768 * 256 * 2);
  bf16* wGhh1 = (bf16*)alloc((size_t)768 * 256 * 2);
  bf16* wSih  = (bf16*)alloc((size_t)2 * 768 * 256 * 2);
  bf16* wShh  = (bf16*)alloc((size_t)2 * 768 * 256 * 2);
  bf16* wCih  = (bf16*)alloc((size_t)2 * 768 * 256 * 2);
  bf16* wChh  = (bf16*)alloc((size_t)2 * 768 * 256 * 2);
  bf16* wSqkv = (bf16*)alloc((size_t)768 * 256 * 2);
  bf16* wCqkv = (bf16*)alloc((size_t)768 * 256 * 2);
  // f32 activations
  float* out1 = (float*)alloc((size_t)M * 256 * 4);
  float* bufA = (float*)alloc((size_t)M * 768 * 4);   // xg / qkv
  float* bufB = (float*)alloc((size_t)M * 256 * 4);   // out0 / comp / gru2-out
  float* bufC = (float*)alloc((size_t)M * 256 * 4);   // gru2-mid / ctx
  float* bufS = (float*)alloc((size_t)64 * 512 * 512 * 4); // scores
  float* hS   = (float*)alloc((size_t)64 * 256 * 4);
  float* hI   = (float*)alloc((size_t)64 * 256 * 4);
  int*   neff = (int*)alloc((size_t)64 * 4);

  auto cvt = [&](const void* src, bf16* dst, int n) {
    cvt_bf16<<<(n + 255) / 256, 256, 0, stream>>>((const float*)src, dst, n);
  };
  cvt(d_in[1],  wGih0, 768 * 1024);
  cvt(d_in[2],  wGhh0, 768 * 256);
  cvt(d_in[5],  wGih1, 768 * 256);
  cvt(d_in[6],  wGhh1, 768 * 256);
  cvt(d_in[9],  wSih, 2 * 768 * 256);
  cvt(d_in[10], wShh, 2 * 768 * 256);
  cvt(d_in[15], wCih, 2 * 768 * 256);
  cvt(d_in[16], wChh, 2 * 768 * 256);
  cvt(d_in[13], wSqkv, 768 * 256);
  cvt(d_in[19], wCqkv, 768 * 256);

  auto gemm_w = [&](const float* A, int K, const bf16* W, const float* bias,
                    float* C, int N, int Mm) {
    dim3 g(Mm / 64, N / 128, 1);
    wmma_gemm<0><<<g, 256, 0, stream>>>(A, K, 0, W, K, 0, bias, C, N, 0, Mm, N, K, 1.f);
  };
  const size_t scan_lds = 16 * 256 * 4 + 16 * 256 * 2 + 16 * 768 * 4 + 768 * 4; // 76800 B = 75 KiB

  // interaction GRU, layer 0 + 1
  gemm_w(x, 1024, wGih0, (const float*)d_in[3], bufA, 768, M);
  gru_scan<<<4, 256, scan_lds, stream>>>(bufA, wGhh0, (const float*)d_in[4], bufB, 512, 64);
  gemm_w(bufB, 256, wGih1, (const float*)d_in[7], bufA, 768, M);
  gru_scan<<<4, 256, scan_lds, stream>>>(bufA, wGhh1, (const float*)d_in[8], out1, 512, 64);

  for (int br = 0; br < 2; ++br) {
    const bf16* wih   = br ? wCih : wSih;
    const bf16* whh   = br ? wChh : wShh;
    const float* bih  = (const float*)d_in[br ? 17 : 11];
    const float* bhh  = (const float*)d_in[br ? 18 : 12];
    const bf16* wqkv  = br ? wCqkv : wSqkv;
    const float* bqkv = (const float*)d_in[br ? 20 : 14];
    float* hout = br ? hI : hS;

    // compact role timesteps (sid = br) into bufB (zero tail)
    fill0<<<(int)(((long long)M * 256 + 255) / 256), 256, 0, stream>>>(bufB, (long long)M * 256);
    compact_kernel<<<64, 256, 0, stream>>>(out1, (const int*)d_in[23], (const int*)d_in[24],
                                           br, bufB, neff);
    // 2-layer role GRU
    gemm_w(bufB, 256, wih, bih, bufA, 768, M);
    gru_scan<<<4, 256, scan_lds, stream>>>(bufA, whh, bhh, bufC, 512, 64);
    gemm_w(bufC, 256, wih + 768 * 256, bih + 768, bufA, 768, M);
    gru_scan<<<4, 256, scan_lds, stream>>>(bufA, whh + 768 * 256, bhh + 768, bufB, 512, 64);
    // fused QKV projection (N = 768 = Q|K|V)
    gemm_w(bufB, 256, wqkv, bqkv, bufA, 768, M);
    // scores = Q K^T / 16, batched over 64
    {
      dim3 g(512 / 64, 512 / 128, 64);
      wmma_gemm<1><<<g, 256, 0, stream>>>(bufA, 768, 512LL * 768,
                                          bufA + 256, 768, 512LL * 768,
                                          nullptr, bufS, 512, 512LL * 512,
                                          512, 512, 256, 0.0625f);
    }
    softmax_rows<<<64 * 512 / 8, 256, 0, stream>>>(bufS, neff, 512);
    // ctx = attn V, batched (B operand is [K=512, N=256] with row stride 768)
    {
      dim3 g(512 / 64, 256 / 128, 64);
      wmma_gemm<2><<<g, 256, 0, stream>>>(bufS, 512, 512LL * 512,
                                          bufA + 512, 768, 512LL * 768,
                                          nullptr, bufC, 256, 512LL * 256,
                                          512, 256, 512, 1.f);
    }
    pool_kernel<<<64, 256, 0, stream>>>(bufC, neff, hout);
  }

  fc_kernel<<<64, 128, 0, stream>>>(hS, hI, out1, (const int*)d_in[23],
                                    (const float*)d_in[21], (const float*)d_in[22],
                                    (float*)d_out);
}
